// AttentionalPooler_76605036692089
// MI455X (gfx1250) — matrix operand order repaired and checked by reading, not compiled, not run on hardware
//
#include <hip/hip_runtime.h>
#include <hip/hip_bf16.h>
#include <cmath>

typedef __attribute__((ext_vector_type(16))) _Float16 v16h;
typedef __attribute__((ext_vector_type(8)))  _Float16 v8h;
typedef __attribute__((ext_vector_type(8)))  float    v8f;

#define B_     32
#define N_     512
#define D_     1024
#define H_     8
#define DH_    64
#define INNER_ 512
#define PROJ_  512
#define MID_   768
#define EPS_   1e-5f

// ---------------------------------------------------------------------------
// WMMA fragment helpers (CDNA5 ISA layouts, wave32)
// A 16x32 f16 (row-major source, leading dim = ld):
//   lane L: row = L%16; elems 0..7  -> K = k0 + (L/16)*8 + e
//                        elems 8..15 -> K = k0 + 16 + (L/16)*8 + (e-8)
// B 32x16 f16 from "Bt" storage (col-major B == row-major Bt[n][k], ldk):
//   lane L: col = L%16; elems 0..7  -> K = k0 + (L/16)*8 + e
//                        elems 8..15 -> K = k0 + 16 + (L/16)*8 + (e-8)
// C/D f32 16x16: vgpr r, lane L -> row = r + 8*(L/16), col = L%16
// ---------------------------------------------------------------------------
__device__ __forceinline__ v16h ld_frag_a(const _Float16* A, int ld, int lane) {
  const _Float16* p = A + (lane & 15) * ld + ((lane >> 4) << 3);
  v8h lo = *(const v8h*)(p);
  v8h hi = *(const v8h*)(p + 16);
  v16h r;
#pragma unroll
  for (int i = 0; i < 8; ++i) { r[i] = lo[i]; r[i + 8] = hi[i]; }
  return r;
}

__device__ __forceinline__ v16h ld_frag_b(const _Float16* Bt, int ldk, int lane) {
  const _Float16* p = Bt + (lane & 15) * ldk + ((lane >> 4) << 3);
  v8h lo = *(const v8h*)(p);
  v8h hi = *(const v8h*)(p + 16);
  v16h r;
#pragma unroll
  for (int i = 0; i < 8; ++i) { r[i] = lo[i]; r[i + 8] = hi[i]; }
  return r;
}

__device__ __forceinline__ v8f wmma_f16(v16h a, v16h b, v8f c) {
  return __builtin_amdgcn_wmma_f32_16x16x32_f16(false, a, false, b, (short)0, c,
                                                false, false);
}

// ---------------------------------------------------------------------------
// Prep: transpose fp32 weight (R x C) -> f16 Bt (C x R), k-contiguous
// ---------------------------------------------------------------------------
__global__ void transpose_to_f16(const float* __restrict__ src,
                                 _Float16* __restrict__ dst, int R, int C) {
  int i = blockIdx.x * blockDim.x + threadIdx.x;
  if (i >= R * C) return;
  int r = i / C, c = i % C;
  dst[c * R + r] = (_Float16)src[i];
}

// ---------------------------------------------------------------------------
// LayerNorm over D=1024, optional pos-encoding add, f16 output
// ---------------------------------------------------------------------------
__global__ __launch_bounds__(256) void ln1024_to_f16(
    const float* __restrict__ x, const float* __restrict__ pos,
    const float* __restrict__ g, _Float16* __restrict__ out, int nmod) {
  int row = blockIdx.x;
  int tid = threadIdx.x;
  const float* xr = x + (size_t)row * D_;
  const float* pr = pos ? pos + (size_t)(row % nmod) * D_ : nullptr;
  float v[4];
  float s = 0.f, s2 = 0.f;
#pragma unroll
  for (int q = 0; q < 4; ++q) {
    int i = tid + q * 256;
    float t = xr[i];
    if (pr) t += pr[i];
    v[q] = t; s += t; s2 += t * t;
  }
#pragma unroll
  for (int o = 16; o > 0; o >>= 1) { s += __shfl_xor(s, o); s2 += __shfl_xor(s2, o); }
  __shared__ float shs[8], shq[8];
  __shared__ float smean, srstd;
  int wv = tid >> 5;
  if ((tid & 31) == 0) { shs[wv] = s; shq[wv] = s2; }
  __syncthreads();
  if (tid == 0) {
    float ts = 0.f, tq = 0.f;
    for (int k = 0; k < 8; ++k) { ts += shs[k]; tq += shq[k]; }
    float mean = ts * (1.0f / D_);
    float var  = tq * (1.0f / D_) - mean * mean;
    smean = mean; srstd = rsqrtf(var + EPS_);
  }
  __syncthreads();
  float mean = smean, rstd = srstd;
#pragma unroll
  for (int q = 0; q < 4; ++q) {
    int i = tid + q * 256;
    out[(size_t)row * D_ + i] = (_Float16)((v[q] - mean) * rstd * g[i]);
  }
}

// ---------------------------------------------------------------------------
// Register-blocked GEMM core: each wave computes a 32x64 macro-tile
// (MT=2 x NT=4 WMMA tiles): 8 wmma per 6 fragment loads per k-chunk.
// A: row-major (M x K) f16, Bt: k-contiguous (N x K) f16.
// ---------------------------------------------------------------------------
#define MT_ 2
#define NT_ 4

template <int LDA, int LDK, int KDIM>
__device__ __forceinline__ void gemm_macro(const _Float16* __restrict__ A,
                                           const _Float16* __restrict__ Bt,
                                           int tm, int tn, int lane,
                                           v8f c[MT_][NT_]) {
  const _Float16* Abase = A + (size_t)(tm * MT_) * 16 * LDA;
  const _Float16* Bbase = Bt + (size_t)(tn * NT_) * 16 * LDK;
  for (int kk = 0; kk < KDIM; kk += 32) {
    v16h a[MT_];
#pragma unroll
    for (int im = 0; im < MT_; ++im)
      a[im] = ld_frag_a(Abase + (size_t)im * 16 * LDA + kk, LDA, lane);
    v16h b[NT_];
#pragma unroll
    for (int in = 0; in < NT_; ++in)
      b[in] = ld_frag_b(Bbase + (size_t)in * 16 * LDK + kk, LDK, lane);
#pragma unroll
    for (int im = 0; im < MT_; ++im)
#pragma unroll
      for (int in = 0; in < NT_; ++in)
        c[im][in] = wmma_f16(a[im], b[in], c[im][in]);
  }
}

// ---------------------------------------------------------------------------
// Q = LN(queries) @ Wq * DH^-0.5   (512x512, K=1024) -> f16
// ---------------------------------------------------------------------------
__global__ __launch_bounds__(256) void gemm_q_kernel(
    const _Float16* __restrict__ A, const _Float16* __restrict__ Bt,
    _Float16* __restrict__ Q) {
  int wid  = (blockIdx.x * blockDim.x + threadIdx.x) >> 5;
  int lane = threadIdx.x & 31;
  const int MACN = (INNER_ / 16) / NT_;  // 8
  int tm = wid / MACN, tn = wid % MACN;
  if (tm >= (N_ / 16) / MT_) return;
  v8f c[MT_][NT_] = {};
  gemm_macro<D_, D_, D_>(A, Bt, tm, tn, lane, c);
  int rb = (lane >> 4) << 3;
#pragma unroll
  for (int im = 0; im < MT_; ++im)
#pragma unroll
    for (int in = 0; in < NT_; ++in) {
      int col = (tn * NT_ + in) * 16 + (lane & 15);
#pragma unroll
      for (int r = 0; r < 8; ++r) {
        int row = (tm * MT_ + im) * 16 + rb + r;
        Q[row * INNER_ + col] = (_Float16)(c[im][in][r] * 0.125f);  // DH^-0.5
      }
    }
}

// ---------------------------------------------------------------------------
// kv = ctx @ Wkv (16384 x 128, K=1024); split into k (B,N,64) and v^T (B,64,N)
// ---------------------------------------------------------------------------
__global__ __launch_bounds__(256) void gemm_kv_kernel(
    const _Float16* __restrict__ ctx, const _Float16* __restrict__ Bt,
    _Float16* __restrict__ kh, _Float16* __restrict__ vt) {
  int wid  = (blockIdx.x * blockDim.x + threadIdx.x) >> 5;
  int lane = threadIdx.x & 31;
  const int MACN = ((2 * DH_) / 16) / NT_;  // 2
  int tm = wid / MACN, tn = wid % MACN;
  if (tm >= ((B_ * N_) / 16) / MT_) return;
  v8f c[MT_][NT_] = {};
  gemm_macro<D_, D_, D_>(ctx, Bt, tm, tn, lane, c);
  int rb = (lane >> 4) << 3;
#pragma unroll
  for (int im = 0; im < MT_; ++im)
#pragma unroll
    for (int in = 0; in < NT_; ++in) {
      int col = (tn * NT_ + in) * 16 + (lane & 15);
#pragma unroll
      for (int r = 0; r < 8; ++r) {
        int row = (tm * MT_ + im) * 16 + rb + r;
        int bb  = row >> 9;
        int nn  = row & (N_ - 1);
        float vv = c[im][in][r];
        if (col < DH_) kh[row * DH_ + col] = (_Float16)vv;
        else           vt[((size_t)bb * DH_ + (col - DH_)) * N_ + nn] = (_Float16)vv;
      }
    }
}

// ---------------------------------------------------------------------------
// Attention: 1 wave per (b, h, 16-query-row tile).
// QK^T (32 tiles, K=64, Q-frags hoisted) -> fp32 sim in LDS -> softmax ->
// f16 probs in LDS -> AV (loop-interchanged: 1 LDS A-frag reused over 4
// d-tiles per k-chunk), row-rescaled by 1/sum at store.
// ---------------------------------------------------------------------------
__global__ __launch_bounds__(32) void attn_kernel(
    const _Float16* __restrict__ Qh, const _Float16* __restrict__ kh,
    const _Float16* __restrict__ vt, _Float16* __restrict__ aout) {
  __shared__ __attribute__((aligned(32))) float    sim[16 * N_];
  __shared__ __attribute__((aligned(32))) _Float16 pa[16 * N_];
  __shared__ float invs[16];

  int lane = threadIdx.x;
  int blk  = blockIdx.x;
  int it = blk & 31;
  int h  = (blk >> 5) & (H_ - 1);
  int b  = blk >> 8;

  const _Float16* Abase = Qh + (it * 16) * INNER_ + h * DH_;
  const _Float16* Kbase = kh + (size_t)b * N_ * DH_;

  // --- sim = Q_h @ k^T ---  (Q-tile fragments are loop-invariant: hoist)
  v16h aq0 = ld_frag_a(Abase, INNER_, lane);
  v16h aq1 = ld_frag_a(Abase + 32, INNER_, lane);
  for (int jt = 0; jt < N_ / 16; ++jt) {
    v8f c = {};
    c = wmma_f16(aq0, ld_frag_b(Kbase + jt * 16 * DH_, DH_, lane), c);
    c = wmma_f16(aq1, ld_frag_b(Kbase + jt * 16 * DH_ + 32, DH_, lane), c);
    int col = jt * 16 + (lane & 15);
    int rb  = (lane >> 4) << 3;
#pragma unroll
    for (int r = 0; r < 8; ++r) sim[(rb + r) * N_ + col] = c[r];
  }
  __syncthreads();

  // --- softmax: 2 lanes per row, 256 cols each ---
  int row  = lane & 15;
  int half = lane >> 4;
  const float* srow = &sim[row * N_ + half * 256];
  float m = -3.0e38f;
  for (int j = 0; j < 256; ++j) m = fmaxf(m, srow[j]);
  m = fmaxf(m, __shfl_xor(m, 16));
  float s = 0.f;
  _Float16* prow = &pa[row * N_ + half * 256];
  for (int j = 0; j < 256; ++j) {
    float e = __expf(srow[j] - m);
    s += e;
    prow[j] = (_Float16)e;
  }
  s += __shfl_xor(s, 16);
  if (lane < 16) invs[lane] = 1.0f / s;
  __syncthreads();

  // --- out = probs @ v ---  (one LDS A-frag feeds all 4 d-tiles per chunk)
  const _Float16* Vbase = vt + (size_t)b * DH_ * N_;
  v8f cav[DH_ / 16] = {};
  for (int kk = 0; kk < N_; kk += 32) {
    v16h a = ld_frag_a(&pa[kk], N_, lane);
#pragma unroll
    for (int dt = 0; dt < DH_ / 16; ++dt) {
      v16h bf = ld_frag_b(Vbase + dt * 16 * N_ + kk, N_, lane);
      cav[dt] = wmma_f16(a, bf, cav[dt]);
    }
  }
  {
    int rb = (lane >> 4) << 3;
#pragma unroll
    for (int dt = 0; dt < DH_ / 16; ++dt) {
      int col = dt * 16 + (lane & 15);
#pragma unroll
      for (int r = 0; r < 8; ++r) {
        int rr = rb + r;
        int gr = b * N_ + it * 16 + rr;
        aout[(size_t)gr * INNER_ + h * DH_ + col] = (_Float16)(cav[dt][r] * invs[rr]);
      }
    }
  }
}

// ---------------------------------------------------------------------------
// out2 = aout @ Wout  (16384 x 1024, K=512) -> fp32
// ---------------------------------------------------------------------------
__global__ __launch_bounds__(256) void gemm_out_kernel(
    const _Float16* __restrict__ A, const _Float16* __restrict__ Bt,
    float* __restrict__ out2) {
  int wid  = (blockIdx.x * blockDim.x + threadIdx.x) >> 5;
  int lane = threadIdx.x & 31;
  const int MACN = (D_ / 16) / NT_;  // 16
  int tm = wid / MACN, tn = wid % MACN;
  if (tm >= ((B_ * N_) / 16) / MT_) return;
  v8f c[MT_][NT_] = {};
  gemm_macro<INNER_, INNER_, INNER_>(A, Bt, tm, tn, lane, c);
  int rb = (lane >> 4) << 3;
#pragma unroll
  for (int im = 0; im < MT_; ++im)
#pragma unroll
    for (int in = 0; in < NT_; ++in) {
      int col = (tn * NT_ + in) * 16 + (lane & 15);
#pragma unroll
      for (int r = 0; r < 8; ++r) {
        int row = (tm * MT_ + im) * 16 + rb + r;
        out2[(size_t)row * D_ + col] = c[im][in][r];
      }
    }
}

// ---------------------------------------------------------------------------
// Per-row mean / rsqrt(var+eps) over D=1024 fp32 rows
// ---------------------------------------------------------------------------
__global__ __launch_bounds__(256) void rowstats1024(const float* __restrict__ x,
                                                    float* __restrict__ rowm,
                                                    float* __restrict__ rowr) {
  int row = blockIdx.x;
  int tid = threadIdx.x;
  const float* xr = x + (size_t)row * D_;
  float s = 0.f, s2 = 0.f;
#pragma unroll
  for (int q = 0; q < 4; ++q) {
    float t = xr[tid + q * 256];
    s += t; s2 += t * t;
  }
#pragma unroll
  for (int o = 16; o > 0; o >>= 1) { s += __shfl_xor(s, o); s2 += __shfl_xor(s2, o); }
  __shared__ float shs[8], shq[8];
  int wv = tid >> 5;
  if ((tid & 31) == 0) { shs[wv] = s; shq[wv] = s2; }
  __syncthreads();
  if (tid == 0) {
    float ts = 0.f, tq = 0.f;
    for (int k = 0; k < 8; ++k) { ts += shs[k]; tq += shq[k]; }
    float mean = ts * (1.0f / D_);
    float var  = tq * (1.0f / D_) - mean * mean;
    rowm[row] = mean;
    rowr[row] = rsqrtf(var + EPS_);
  }
}

// ---------------------------------------------------------------------------
// emb[b][d] = mean_n LN(out2)[b,n,d] * g[d]
// ---------------------------------------------------------------------------
__global__ __launch_bounds__(128) void emb_kernel(
    const float* __restrict__ out2, const float* __restrict__ rowm,
    const float* __restrict__ rowr, const float* __restrict__ g,
    float* __restrict__ emb) {
  int b = blockIdx.x >> 3;
  int d = (blockIdx.x & 7) * 128 + threadIdx.x;
  float acc = 0.f;
  for (int n = 0; n < N_; ++n) {
    int r = b * N_ + n;
    acc += (out2[(size_t)r * D_ + d] - rowm[r]) * rowr[r];
  }
  emb[b * D_ + d] = acc * g[d] * (1.0f / N_);
}

// ---------------------------------------------------------------------------
// MLP head (tiny): fp32 VALU, exact GELU
// ---------------------------------------------------------------------------
__global__ __launch_bounds__(256) void mlp1_kernel(
    const float* __restrict__ emb, const float* __restrict__ W1,
    const float* __restrict__ b1, float* __restrict__ mid) {
  int i = blockIdx.x * blockDim.x + threadIdx.x;
  if (i >= B_ * MID_) return;
  int b = i / MID_, m = i % MID_;
  const float* e = emb + b * D_;
  float acc = b1[m];
  for (int k = 0; k < D_; ++k) acc += e[k] * W1[k * MID_ + m];
  mid[i] = 0.5f * acc * (1.0f + erff(acc * 0.70710678118654752f));
}

__global__ __launch_bounds__(256) void mlp2_kernel(
    const float* __restrict__ mid, const float* __restrict__ W2,
    const float* __restrict__ b2, float* __restrict__ pred) {
  int i = blockIdx.x * blockDim.x + threadIdx.x;
  if (i >= B_ * PROJ_) return;
  int b = i / PROJ_, p = i % PROJ_;
  const float* mrow = mid + b * MID_;
  float acc = b2[p];
  for (int k = 0; k < MID_; ++k) acc += mrow[k] * W2[k * PROJ_ + p];
  pred[i] = acc;
}

// ---------------------------------------------------------------------------
extern "C" void kernel_launch(void* const* d_in, const int* in_sizes, int n_in,
                              void* d_out, int out_size, void* d_ws, size_t ws_size,
                              hipStream_t stream) {
  (void)in_sizes; (void)n_in; (void)out_size; (void)ws_size;
  const float* x    = (const float*)d_in[0];
  const float* pos  = (const float*)d_in[1];
  const float* qrs  = (const float*)d_in[2];
  const float* lnq  = (const float*)d_in[3];
  const float* lnc  = (const float*)d_in[4];
  const float* Wq   = (const float*)d_in[5];
  const float* Wkv  = (const float*)d_in[6];
  const float* Wout = (const float*)d_in[7];
  const float* lnp  = (const float*)d_in[8];
  const float* W1   = (const float*)d_in[9];
  const float* b1   = (const float*)d_in[10];
  const float* W2   = (const float*)d_in[11];
  const float* b2   = (const float*)d_in[12];
  float* pred = (float*)d_out;

  char* w = (char*)d_ws;
  size_t off = 0;
  auto alloc = [&](size_t bytes) -> void* {
    void* p = w + off;
    off += (bytes + 255) & ~(size_t)255;
    return p;
  };
  _Float16* qn_h  = (_Float16*)alloc((size_t)N_ * D_ * 2);
  _Float16* WqT   = (_Float16*)alloc((size_t)INNER_ * D_ * 2);
  _Float16* WkvT  = (_Float16*)alloc((size_t)2 * DH_ * D_ * 2);
  _Float16* WoutT = (_Float16*)alloc((size_t)D_ * INNER_ * 2);
  _Float16* Q_h   = (_Float16*)alloc((size_t)N_ * INNER_ * 2);
  _Float16* ctx_h = (_Float16*)alloc((size_t)B_ * N_ * D_ * 2);
  _Float16* k_h   = (_Float16*)alloc((size_t)B_ * N_ * DH_ * 2);
  _Float16* v_t   = (_Float16*)alloc((size_t)B_ * DH_ * N_ * 2);
  _Float16* aout  = (_Float16*)alloc((size_t)B_ * N_ * INNER_ * 2);
  float* out2     = (float*)alloc((size_t)B_ * N_ * D_ * 4);
  float* rowm     = (float*)alloc((size_t)B_ * N_ * 4);
  float* rowr     = (float*)alloc((size_t)B_ * N_ * 4);
  float* emb      = (float*)alloc((size_t)B_ * D_ * 4);
  float* mid      = (float*)alloc((size_t)B_ * MID_ * 4);

  // weight prep (f16, transposed to k-contiguous "Bt")
  transpose_to_f16<<<(D_ * INNER_ + 255) / 256, 256, 0, stream>>>(Wq, WqT, D_, INNER_);
  transpose_to_f16<<<(D_ * 2 * DH_ + 255) / 256, 256, 0, stream>>>(Wkv, WkvT, D_, 2 * DH_);
  transpose_to_f16<<<(INNER_ * D_ + 255) / 256, 256, 0, stream>>>(Wout, WoutT, INNER_, D_);

  // layernorms (queries once; ctx = LN(x + pos))
  ln1024_to_f16<<<N_, 256, 0, stream>>>(qrs, nullptr, lnq, qn_h, 1);
  ln1024_to_f16<<<B_ * N_, 256, 0, stream>>>(x, pos, lnc, ctx_h, N_);

  // Q (batch-independent): 16x8 macro-tiles = 128 waves
  gemm_q_kernel<<<(128 * 32) / 256, 256, 0, stream>>>(qn_h, WqT, Q_h);

  // kv: 512x2 macro-tiles = 1024 waves
  gemm_kv_kernel<<<(1024 * 32) / 256, 256, 0, stream>>>(ctx_h, WkvT, k_h, v_t);

  // attention: one wave per (b, h, query-row tile)
  attn_kernel<<<B_ * H_ * (N_ / 16), 32, 0, stream>>>(Q_h, k_h, v_t, aout);

  // output projection: 512x16 macro-tiles = 8192 waves
  gemm_out_kernel<<<(8192 * 32) / 256, 256, 0, stream>>>(aout, WoutT, out2);

  // post-LN stats + pooled embedding
  rowstats1024<<<B_ * N_, 256, 0, stream>>>(out2, rowm, rowr);
  emb_kernel<<<B_ * 8, 128, 0, stream>>>(out2, rowm, rowr, lnp, emb);

  // MLP head
  mlp1_kernel<<<(B_ * MID_ + 255) / 256, 256, 0, stream>>>(emb, W1, b1, mid);
  mlp2_kernel<<<(B_ * PROJ_ + 255) / 256, 256, 0, stream>>>(mid, W2, b2, pred);
}
